// AJRNN_48249662603445
// MI455X (gfx1250) — matrix operand order repaired and checked by reading, not compiled, not run on hardware
//
#include <hip/hip_runtime.h>
#include <hip/hip_bf16.h>

// ---------------------------------------------------------------------------
// CDNA5 (gfx1250) AJRNN: 2-layer LSTM recurrence with WMMA bf16 tensor ops.
// One wave owns all 4 gates of a 16x16 tile -> A loaded once per K-chunk,
// gate activations fused in registers (no LDS, no barriers), hardware
// transcendentals (v_exp_f32 / v_rcp_f32) for the serial-critical epilogue.
// ---------------------------------------------------------------------------

typedef __attribute__((ext_vector_type(16))) __bf16        v16bf;
typedef __attribute__((ext_vector_type(8)))  float         v8f;
typedef __attribute__((ext_vector_type(4)))  unsigned int  u32x4;

#define B_DIM 256
#define T_DIM 256
#define D_DIM 32
#define H_DIM 512
#define C_DIM 10
#define MISSING_VAL 128.0f

union ABu { v16bf v; u32x4 q[2]; };

// A operand: 16x32 bf16 tile from row-major [rows x lda] bf16 matrix.
// Lane L (<16): row = mt*16+L, K = kbase+{0..15}
// Lane L (>=16): row = mt*16+(L-16), K = kbase+16+{0..15}
static __device__ __forceinline__ v16bf load_a(const __bf16* base, int row, int lda,
                                               int kbase, int lane) {
  int khalf = (lane & 16) ? 16 : 0;
  const __bf16* p = base + (size_t)row * lda + kbase + khalf;
  ABu a;
  a.q[0] = *(const u32x4*)(p);
  a.q[1] = *(const u32x4*)(p + 8);
  return a.v;
}

// B operand: pre-packed so each lane reads 16 contiguous bf16 (32 bytes).
static __device__ __forceinline__ v16bf load_bp(const __bf16* pk, int ntiles,
                                                int kt, int nt, int lane) {
  const __bf16* p = pk + (((size_t)kt * ntiles + nt) * 32 + lane) * 16;
  ABu b;
  b.q[0] = *(const u32x4*)(p);
  b.q[1] = *(const u32x4*)(p + 8);
  return b.v;
}

static __device__ __forceinline__ v8f wmma_bf16(v16bf a, v16bf b, v8f c) {
  return __builtin_amdgcn_wmma_f32_16x16x32_bf16(false, a, false, b,
                                                 (short)0, c, false, false);
}

// Fast activations on the hardware transcendental pipe (v_exp_f32, v_rcp_f32).
static __device__ __forceinline__ float fast_sigmoid(float x) {
  return __builtin_amdgcn_rcpf(1.0f + __expf(-x));   // exp(inf)->inf, rcp(inf)->0
}
static __device__ __forceinline__ float fast_tanh(float x) {
  // tanh(x) = 1 - 2/(exp(2x)+1); saturates to +/-1, no NaN at overflow.
  return 1.0f - 2.0f * __builtin_amdgcn_rcpf(__expf(2.0f * x) + 1.0f);
}

// ---------------------------------------------------------------------------
// Weight packing: f32 row-major [K x Nsrc] -> bf16 WMMA B-operand layout,
// N padded to Npad (multiple of 16) with zeros.
// ---------------------------------------------------------------------------
__global__ void pack_b_kernel(const float* __restrict__ W, __bf16* __restrict__ dst,
                              int K, int Nsrc, int Npad) {
  int idx = blockIdx.x * blockDim.x + threadIdx.x;
  int total = K * Npad;
  if (idx >= total) return;
  int k = idx / Npad, n = idx % Npad;
  float v = (n < Nsrc) ? W[(size_t)k * Nsrc + n] : 0.0f;
  int kt = k >> 5, kin = k & 31, nt = n >> 4, nin = n & 15;
  int lane = nin + (kin & 16);   // kin>=16 -> upper half-wave
  int j = kin & 15;
  int ntiles = Npad >> 4;
  dst[(((size_t)kt * ntiles + nt) * 32 + lane) * 16 + j] = (__bf16)v;
}

// Zero-init cell states and initial hidden ping buffers.
__global__ void init_state_kernel(float* c0, float* c1, __bf16* h0z, __bf16* h1z) {
  int i = blockIdx.x * blockDim.x + threadIdx.x;
  if (i < B_DIM * H_DIM) {
    c0[i] = 0.0f; c1[i] = 0.0f;
    h0z[i] = (__bf16)0.0f; h1z[i] = (__bf16)0.0f;
  }
}

// t = 0: cur = x[:,0,:] verbatim (reference uses raw first frame).
__global__ void cvt0_kernel(const float* __restrict__ x, __bf16* __restrict__ curb) {
  int i = blockIdx.x * blockDim.x + threadIdx.x;
  if (i >= B_DIM * D_DIM) return;
  int b = i >> 5, d = i & 31;
  curb[i] = (__bf16)x[(size_t)b * T_DIM * D_DIM + d];
}

// ---------------------------------------------------------------------------
// t >= 1: pred = h1_prev @ Wproj + bproj  (256x32, K=512, WMMA);
// cur = where(x[:,t]==MISSING, pred, x[:,t]); pred written directly into the
// `prediction` output (row b*(T-1)+(t-1)).
// grid (16), block 32: one wave per batch tile, both D-tiles (A reused).
// ---------------------------------------------------------------------------
__global__ __launch_bounds__(32) void cur_pred_kernel(
    const float* __restrict__ x, int t,
    const __bf16* __restrict__ h1_prev, const __bf16* __restrict__ Wprojp,
    const float* __restrict__ bproj,
    __bf16* __restrict__ curb, float* __restrict__ pred_out) {
  int lane = threadIdx.x;
  int mt = blockIdx.x;                 // batch tile 0..15
  v8f acc[2] = {{}, {}};
  int arow = mt * 16 + (lane & 15);
  for (int kt = 0; kt < H_DIM / 32; ++kt) {
    v16bf a = load_a(h1_prev, arow, H_DIM, kt * 32, lane);
#pragma unroll
    for (int nt = 0; nt < 2; ++nt) {
      v16bf b = load_bp(Wprojp, D_DIM / 16, kt, nt, lane);
      acc[nt] = wmma_bf16(a, b, acc[nt]);
    }
  }
  int m0 = (lane & 16) ? 8 : 0;
#pragma unroll
  for (int nt = 0; nt < 2; ++nt) {
    int col = nt * 16 + (lane & 15);
    float bv = bproj[col];
#pragma unroll
    for (int r = 0; r < 8; ++r) {
      int row = mt * 16 + m0 + r;                            // batch index
      float p = acc[nt][r] + bv;
      float xv = x[((size_t)row * T_DIM + t) * D_DIM + col];
      float cv = (xv == MISSING_VAL) ? p : xv;
      curb[row * D_DIM + col] = (__bf16)cv;
      pred_out[((size_t)row * (T_DIM - 1) + (t - 1)) * D_DIM + col] = p;
    }
  }
}

// ---------------------------------------------------------------------------
// One LSTM layer step: z = A1@W1 + A2@W2 + b; gate activations; update c, h.
// grid (16, 8), block 128 = 4 independent waves; wave w handles hidden tile
// nt = blockIdx.y*4 + w and owns ALL FOUR gate accumulators for its 16x16
// (batch x hidden) tile: A loaded once per K-chunk feeds 4 WMMAs, and the
// LSTM elementwise update runs entirely in registers.
// ---------------------------------------------------------------------------
__global__ __launch_bounds__(128) void lstm_layer_kernel(
    const __bf16* __restrict__ A1, int K1, const __bf16* __restrict__ W1p,
    const __bf16* __restrict__ A2, const __bf16* __restrict__ W2p,
    const float* __restrict__ bias,
    float* __restrict__ c,               // [B,H] in/out (tile-owned, in place)
    __bf16* __restrict__ h_bf,           // [B,H] out (ping-pong, != read buf)
    float* __restrict__ h_f32) {         // optional f32 copy (layer 1)
  int lane = threadIdx.x & 31;
  int wv = threadIdx.x >> 5;
  int mt = blockIdx.x;                   // batch tile 0..15
  int nt = blockIdx.y * 4 + wv;          // hidden tile within H: 0..31
  const int ntB = (4 * H_DIM) / 16;      // 128 column tiles in packed W

  v8f acc[4] = {{}, {}, {}, {}};         // gates i, f, g, o
  int arow = mt * 16 + (lane & 15);

  for (int kt = 0; kt < K1 / 32; ++kt) {
    v16bf a = load_a(A1, arow, K1, kt * 32, lane);
#pragma unroll
    for (int g = 0; g < 4; ++g) {
      v16bf b = load_bp(W1p, ntB, kt, g * (H_DIM / 16) + nt, lane);
      acc[g] = wmma_bf16(a, b, acc[g]);
    }
  }
  for (int kt = 0; kt < H_DIM / 32; ++kt) {
    v16bf a = load_a(A2, arow, H_DIM, kt * 32, lane);
#pragma unroll
    for (int g = 0; g < 4; ++g) {
      v16bf b = load_bp(W2p, ntB, kt, g * (H_DIM / 16) + nt, lane);
      acc[g] = wmma_bf16(a, b, acc[g]);
    }
  }

  int ncol = nt * 16 + (lane & 15);
  float bi = bias[0 * H_DIM + ncol];
  float bf = bias[1 * H_DIM + ncol];
  float bg = bias[2 * H_DIM + ncol];
  float bo = bias[3 * H_DIM + ncol];
  int m0 = (lane & 16) ? 8 : 0;
#pragma unroll
  for (int r = 0; r < 8; ++r) {
    int row = mt * 16 + m0 + r;
    size_t idx = (size_t)row * H_DIM + ncol;
    float iv = fast_sigmoid(acc[0][r] + bi);
    float fv = fast_sigmoid(acc[1][r] + bf);
    float gv = fast_tanh(acc[2][r] + bg);
    float ov = fast_sigmoid(acc[3][r] + bo);
    float cn = fv * c[idx] + iv * gv;
    float hn = ov * fast_tanh(cn);
    c[idx] = cn;
    h_bf[idx] = (__bf16)hn;
    if (h_f32) h_f32[idx] = hn;
  }
}

// ---------------------------------------------------------------------------
// Epilogue: last = h1 (f32 copy); logits = last @ Wcls + bcls (tiny).
// ---------------------------------------------------------------------------
__global__ __launch_bounds__(256) void final_kernel(
    const float* __restrict__ h1f, const float* __restrict__ Wcls,
    const float* __restrict__ bcls,
    float* __restrict__ logits_out, float* __restrict__ last_out) {
  int b = blockIdx.x;
  for (int n = threadIdx.x; n < H_DIM; n += 256)
    last_out[(size_t)b * H_DIM + n] = h1f[(size_t)b * H_DIM + n];
  if (threadIdx.x < C_DIM) {
    int cc = threadIdx.x;
    float s = bcls[cc];
    for (int n = 0; n < H_DIM; ++n)
      s += h1f[(size_t)b * H_DIM + n] * Wcls[n * C_DIM + cc];
    logits_out[b * C_DIM + cc] = s;
  }
}

// ---------------------------------------------------------------------------
extern "C" void kernel_launch(void* const* d_in, const int* in_sizes, int n_in,
                              void* d_out, int out_size, void* d_ws, size_t ws_size,
                              hipStream_t stream) {
  const float* x     = (const float*)d_in[0];   // (B,T,D)
  const float* Wx0   = (const float*)d_in[1];   // (D,4H)
  const float* U0    = (const float*)d_in[2];   // (H,4H)
  const float* b0    = (const float*)d_in[3];   // (4H,)
  const float* Wx1   = (const float*)d_in[4];   // (H,4H)
  const float* U1    = (const float*)d_in[5];   // (H,4H)
  const float* b1    = (const float*)d_in[6];   // (4H,)
  const float* Wproj = (const float*)d_in[7];   // (H,D)
  const float* bproj = (const float*)d_in[8];   // (D,)
  const float* Wcls  = (const float*)d_in[9];   // (H,C)
  const float* bcls  = (const float*)d_in[10];  // (C,)

  float* out = (float*)d_out;
  float* pred_out   = out;                                           // (B*(T-1), D)
  float* logits_out = out + (size_t)B_DIM * (T_DIM - 1) * D_DIM;     // (B, C)
  float* last_out   = logits_out + (size_t)B_DIM * C_DIM;            // (B, H)

  // --- carve workspace ---
  char* ws = (char*)d_ws;
  size_t off = 0;
  auto carve = [&](size_t bytes) -> void* {
    void* p = ws + off;
    off = (off + bytes + 255) & ~(size_t)255;
    return p;
  };
  __bf16* Wx0p   = (__bf16*)carve((size_t)D_DIM * 4 * H_DIM * 2);
  __bf16* U0p    = (__bf16*)carve((size_t)H_DIM * 4 * H_DIM * 2);
  __bf16* Wx1p   = (__bf16*)carve((size_t)H_DIM * 4 * H_DIM * 2);
  __bf16* U1p    = (__bf16*)carve((size_t)H_DIM * 4 * H_DIM * 2);
  __bf16* Wprojp = (__bf16*)carve((size_t)H_DIM * D_DIM * 2);
  __bf16* h0buf[2], *h1buf[2];
  h0buf[0] = (__bf16*)carve((size_t)B_DIM * H_DIM * 2);
  h0buf[1] = (__bf16*)carve((size_t)B_DIM * H_DIM * 2);
  h1buf[0] = (__bf16*)carve((size_t)B_DIM * H_DIM * 2);
  h1buf[1] = (__bf16*)carve((size_t)B_DIM * H_DIM * 2);
  float* c0    = (float*)carve((size_t)B_DIM * H_DIM * 4);
  float* c1    = (float*)carve((size_t)B_DIM * H_DIM * 4);
  float* h1f32 = (float*)carve((size_t)B_DIM * H_DIM * 4);
  __bf16* curb = (__bf16*)carve((size_t)B_DIM * D_DIM * 2);
  (void)ws_size; (void)n_in; (void)in_sizes; (void)out_size;

  // --- pack weights into WMMA B-operand layout (every call; graph-safe) ---
  auto packN = [&](const float* src, __bf16* dst, int K, int Nsrc, int Npad) {
    int total = K * Npad;
    pack_b_kernel<<<(total + 255) / 256, 256, 0, stream>>>(src, dst, K, Nsrc, Npad);
  };
  packN(Wx0,   Wx0p,   D_DIM, 4 * H_DIM, 4 * H_DIM);
  packN(U0,    U0p,    H_DIM, 4 * H_DIM, 4 * H_DIM);
  packN(Wx1,   Wx1p,   H_DIM, 4 * H_DIM, 4 * H_DIM);
  packN(U1,    U1p,    H_DIM, 4 * H_DIM, 4 * H_DIM);
  packN(Wproj, Wprojp, H_DIM, D_DIM,     D_DIM);

  init_state_kernel<<<(B_DIM * H_DIM + 255) / 256, 256, 0, stream>>>(
      c0, c1, h0buf[1], h1buf[1]);

  // --- recurrence: 3 kernels per timestep, serialized on the stream ---
  dim3 lstmGrid(B_DIM / 16, H_DIM / 16 / 4);   // 16 x 8, 4 waves/block
  for (int t = 0; t < T_DIM; ++t) {
    int rd = (t + 1) & 1, wr = t & 1;          // ping-pong: read rd, write wr
    if (t == 0) {
      cvt0_kernel<<<(B_DIM * D_DIM + 255) / 256, 256, 0, stream>>>(x, curb);
    } else {
      cur_pred_kernel<<<B_DIM / 16, 32, 0, stream>>>(
          x, t, h1buf[rd], Wprojp, bproj, curb, pred_out);
    }
    // layer 0: z0 = cur @ Wx0 + h0 @ U0 + b0
    lstm_layer_kernel<<<lstmGrid, 128, 0, stream>>>(
        curb, D_DIM, Wx0p, h0buf[rd], U0p, b0, c0, h0buf[wr], (float*)nullptr);
    // layer 1: z1 = h0_new @ Wx1 + h1 @ U1 + b1
    lstm_layer_kernel<<<lstmGrid, 128, 0, stream>>>(
        h0buf[wr], H_DIM, Wx1p, h1buf[rd], U1p, b1, c1, h1buf[wr], h1f32);
  }

  final_kernel<<<B_DIM, 256, 0, stream>>>(h1f32, Wcls, bcls, logits_out, last_out);
}
